// MSDeformableAttention_16088947490924
// MI455X (gfx1250) — matrix operand order repaired and checked by reading, not compiled, not run on hardware
//
#include <hip/hip_runtime.h>

typedef __attribute__((ext_vector_type(16))) _Float16 v16h;
typedef __attribute__((ext_vector_type(8)))  float    v8f;

#define BS  4
#define LQ  8192
#define LV  21504
#define DIM 256
#define NH  8
#define NL  3
#define NP  4
#define HD  32
#define GK  256                 // K is 256 for all four GEMMs
#define APAD 4                  // row stride 1040 B -> +4 banks/row, conflict-free
#define ASTRIDE (GK + APAD)     // 260 floats

typedef __attribute__((address_space(3))) char lds_byte;

// ---------------------------------------------------------------------------
// Weight pre-pass: W[K=256, N] fp32 row-major  ->  Bt[N, 256] f16.
// Bt element order along k matches the WMMA B-fragment element order, so a
// fragment is one contiguous 32 B vector load.
// ---------------------------------------------------------------------------
__global__ void transpose_cvt_weight_kernel(const float* __restrict__ W,
                                            _Float16* __restrict__ Bt,
                                            int N) {
  int t = blockIdx.x * blockDim.x + threadIdx.x;
  if (t >= N * GK) return;
  int n = t / GK;
  int k = t - n * GK;                         // consecutive t -> contiguous writes
  Bt[(size_t)n * GK + k] = (_Float16)W[(size_t)k * N + n];
}

// ---------------------------------------------------------------------------
// WMMA GEMM:  C[M,N] = A[M,256](fp32) * Bt[N,256](f16, pre-transposed) + bias
// grid.x = M/16 blocks, 256 threads (8 waves).
//  - 16x256 fp32 A panel async-copied global->LDS once per block
//    (global_load_async_to_lds_b128 + s_wait_asynccnt, ASYNCcnt-tracked).
//  - wave w computes the 16x32 slab [w*32, w*32+32): A fragment re-read from
//    LDS (ds_load + cvt), B fragment = two contiguous b128 loads of f16.
//  - 2 x v_wmma_f32_16x16x32_f16 per K-step. N%32==0 and N/16 even
//    (256/192/96 all qualify), so no lane-divergent branches around WMMA.
// ---------------------------------------------------------------------------
__global__ void __launch_bounds__(256)
wmma_gemm_bias_kernel(const float* __restrict__ A,
                      const _Float16* __restrict__ Bt,
                      const float* __restrict__ bias,
                      float* __restrict__ C, int N) {
  __shared__ float As[16 * ASTRIDE];

  const int tm   = blockIdx.x;          // 16-row M tile
  const int t    = threadIdx.x;
  const int lane = t & 31;
  const int wave = t >> 5;              // 0..7
  const int l16  = lane & 15;
  const int hb   = lane >> 4;           // half-wave select

  // ---- async stage: A[tm*16 .. +15][0..255] fp32 -> LDS (16 KB) ----
  {
    const int r  = t >> 4;              // row 0..15
    const int c0 = (t & 15) * 16;       // 16 floats (64 B) per thread
    const float* gsrc = A + ((size_t)tm * 16 + r) * GK + c0;
    // wave-relative LDS byte offset via explicit addrspace(3) ptrtoint
    const unsigned lbase =
        (unsigned)(size_t)(lds_byte*)&As[r * ASTRIDE + c0];
#pragma unroll
    for (int j = 0; j < 4; ++j) {
      unsigned     la = lbase + j * 16;
      const float* ga = gsrc + j * 4;
      asm volatile("global_load_async_to_lds_b128 %0, %1, off"
                   :: "v"(la), "v"(ga) : "memory");
    }
    asm volatile("s_wait_asynccnt 0x0" ::: "memory");
  }
  __syncthreads();

  const int tilesN = N >> 4;            // 16 / 12 / 6 (always even)
  const int tn0    = wave * 2;          // 2 N-tiles per wave
  if (tn0 >= tilesN) return;            // whole wave exits; EXEC stays full

  const int col0 = tn0 * 16 + l16;      // this lane's two B/C columns
  const int col1 = col0 + 16;
  const _Float16* b0p = Bt + (size_t)col0 * GK + hb * 16;
  const _Float16* b1p = Bt + (size_t)col1 * GK + hb * 16;

  v8f acc0 = {}, acc1 = {};
#pragma unroll
  for (int k0 = 0; k0 < GK; k0 += 32) {
    // A fragment from LDS: lane row l16; K = k0 + 16*hb + e
    const float* arow = &As[l16 * ASTRIDE + k0 + hb * 16];
    v16h a;
#pragma unroll
    for (int e = 0; e < 8; ++e) {
      a[e]     = (_Float16)arow[e];
      a[e + 8] = (_Float16)arow[8 + e];
    }
    // B fragments: 16 contiguous halves (element e -> K = k0 + 16*hb + e)
    v16h b0 = *(const v16h*)(b0p + k0);
    v16h b1 = *(const v16h*)(b1p + k0);
    acc0 = __builtin_amdgcn_wmma_f32_16x16x32_f16(
        false, a, false, b0, (short)0, acc0, false, false);
    acc1 = __builtin_amdgcn_wmma_f32_16x16x32_f16(
        false, a, false, b1, (short)0, acc1, false, false);
  }

  // ---- C store: VGPR i -> row tm*16 + 8*hb + i ----
  const float bv0 = bias[col0];
  const float bv1 = bias[col1];
  float* crow = C + (size_t)(tm * 16 + hb * 8) * N;
#pragma unroll
  for (int i = 0; i < 8; ++i) {
    crow[i * N + col0] = acc0[i] + bv0;
    crow[i * N + col1] = acc1[i] + bv1;
  }
}

// ---------------------------------------------------------------------------
// Softmax over the 12 (NL*NP) logits per (b,q,h); in-place.
// ---------------------------------------------------------------------------
__global__ void softmax12_kernel(float* __restrict__ attn) {
  int t = blockIdx.x * blockDim.x + threadIdx.x;
  if (t >= BS * LQ * NH) return;
  int bq = t >> 3;
  int h  = t & 7;
  float* p = attn + (size_t)bq * (NH * NL * NP) + h * (NL * NP);
  float m = p[0];
#pragma unroll
  for (int i = 1; i < NL * NP; ++i) m = fmaxf(m, p[i]);
  float s = 0.f;
  float e[NL * NP];
#pragma unroll
  for (int i = 0; i < NL * NP; ++i) { e[i] = __expf(p[i] - m); s += e[i]; }
  float inv = 1.f / s;
#pragma unroll
  for (int i = 0; i < NL * NP; ++i) p[i] = e[i] * inv;
}

// ---------------------------------------------------------------------------
// Deformable sampling. One wave per (b,q,h); lane = channel c (HD==32==wave).
// Faithful to reference: per level the (hw,HD) block is row-major
// reinterpreted as (HD,hw): element (c,s) lives at
// v[(b*LV + start + (c*hw+s)/HD)*DIM + h*HD + (c*hw+s)%HD].
// Bilinear: align_corners=False, zero padding via validity mask.
// ---------------------------------------------------------------------------
__global__ void deform_sample_kernel(const float* __restrict__ v,
                                     const float* __restrict__ off,
                                     const float* __restrict__ aw,
                                     const float* __restrict__ refpts,
                                     float* __restrict__ out) {
  const int lane  = threadIdx.x & 31;        // channel c
  const int gwave = (blockIdx.x * blockDim.x + threadIdx.x) >> 5;
  if (gwave >= BS * LQ * NH) return;
  const int b  = gwave / (LQ * NH);
  const int r  = gwave - b * (LQ * NH);
  const int q  = r / NH;
  const int h  = r - q * NH;

  const int lvl_h[NL]     = {128, 64, 32};
  const int lvl_w[NL]     = {128, 64, 32};
  const int lvl_start[NL] = {0, 16384, 20480};

  const size_t bq   = (size_t)b * LQ + q;
  const float  refx = refpts[bq * 4 + 0];
  const float  refy = refpts[bq * 4 + 1];
  const float* offp = off + bq * (NH * NL * NP * 2) + (size_t)h * (NL * NP * 2);
  const float* awp  = aw  + bq * (NH * NL * NP)     + (size_t)h * (NL * NP);

  float acc = 0.f;
#pragma unroll
  for (int l = 0; l < NL; ++l) {
    const int hh = lvl_h[l], ww = lvl_w[l];
    const int hw = hh * ww;
    const size_t vbase = ((size_t)b * LV + lvl_start[l]) * DIM + h * HD;
    const size_t crow  = (size_t)lane * hw;   // c*hw of the reinterpretation
#pragma unroll
    for (int p = 0; p < NP; ++p) {
      const float ox = offp[(l * NP + p) * 2 + 0];
      const float oy = offp[(l * NP + p) * 2 + 1];
      const float px = fminf(fmaxf(refx + ox, 0.f), 1.f);
      const float py = fminf(fmaxf(refy + oy, 0.f), 1.f);
      const float gx = px * 2.f - 1.f;
      const float gy = py * 2.f - 1.f;
      const float x = ((gx + 1.f) * ww - 1.f) * 0.5f;
      const float y = ((gy + 1.f) * hh - 1.f) * 0.5f;
      const float x0f = floorf(x), y0f = floorf(y);
      const float wx1 = x - x0f, wy1 = y - y0f;
      const int x0 = (int)x0f, y0 = (int)y0f;

      const int   cx[4] = {x0, x0 + 1, x0,     x0 + 1};
      const int   cy[4] = {y0, y0,     y0 + 1, y0 + 1};
      const float cw[4] = {(1.f - wx1) * (1.f - wy1), wx1 * (1.f - wy1),
                           (1.f - wx1) * wy1,         wx1 * wy1};
      float samp = 0.f;
#pragma unroll
      for (int k = 0; k < 4; ++k) {
        const int xi = cx[k], yi = cy[k];
        const bool valid = (xi >= 0) & (xi < ww) & (yi >= 0) & (yi < hh);
        const int xc = min(max(xi, 0), ww - 1);
        const int yc = min(max(yi, 0), hh - 1);
        const int s  = yc * ww + xc;
        const size_t flat = crow + (size_t)s;          // c*hw + s
        const size_t srow = flat >> 5;                 // / HD
        const int    ch   = (int)(flat & 31);          // % HD
        const float  val  = v[vbase + srow * DIM + ch];
        samp += val * (valid ? cw[k] : 0.f);
      }
      acc += awp[l * NP + p] * samp;
    }
  }
  out[(bq * NH + h) * HD + lane] = acc;   // (b,q,h,c) == (bs*Lq, D) layout
}

// ---------------------------------------------------------------------------
extern "C" void kernel_launch(void* const* d_in, const int* in_sizes, int n_in,
                              void* d_out, int out_size, void* d_ws, size_t ws_size,
                              hipStream_t stream) {
  const float* query  = (const float*)d_in[0];   // (4,8192,256)
  const float* refpts = (const float*)d_in[1];   // (4,8192,4)
  const float* value  = (const float*)d_in[2];   // (4,21504,256)
  const float* W_off  = (const float*)d_in[3];   // (256,192)
  const float* b_off  = (const float*)d_in[4];   // (192)
  const float* W_attn = (const float*)d_in[5];   // (256,96)
  const float* b_attn = (const float*)d_in[6];   // (96)
  const float* W_v    = (const float*)d_in[7];   // (256,256)
  const float* b_v    = (const float*)d_in[8];   // (256)
  const float* W_out  = (const float*)d_in[9];   // (256,256)
  const float* b_out  = (const float*)d_in[10];  // (256)
  float* outp = (float*)d_out;

  // workspace carve-up: fp32 tensors, then f16 transposed weights (~160 MB)
  float* ws   = (float*)d_ws;
  float* v    = ws;                                         // 86016*256 f32
  float* off  = v    + (size_t)BS * LV * DIM;               // 32768*192 f32
  float* attn = off  + (size_t)BS * LQ * NH * NL * NP * 2;  // 32768*96  f32
  float* samp = attn + (size_t)BS * LQ * NH * NL * NP;      // 32768*256 f32
  _Float16* Bt_v    = (_Float16*)(samp + (size_t)BS * LQ * DIM);
  _Float16* Bt_off  = Bt_v    + (size_t)DIM * GK;           // 256x256 f16
  _Float16* Bt_attn = Bt_off  + (size_t)(NH * NL * NP * 2) * GK;
  _Float16* Bt_out  = Bt_attn + (size_t)(NH * NL * NP) * GK;

  const int TPB = 256;  // 8 waves per block
  const int Noff  = NH * NL * NP * 2;  // 192
  const int Nattn = NH * NL * NP;      // 96

  // 0) pre-transpose + f16-convert the four weight matrices
  transpose_cvt_weight_kernel<<<(DIM   * GK + TPB - 1) / TPB, TPB, 0, stream>>>(W_v,    Bt_v,    DIM);
  transpose_cvt_weight_kernel<<<(Noff  * GK + TPB - 1) / TPB, TPB, 0, stream>>>(W_off,  Bt_off,  Noff);
  transpose_cvt_weight_kernel<<<(Nattn * GK + TPB - 1) / TPB, TPB, 0, stream>>>(W_attn, Bt_attn, Nattn);
  transpose_cvt_weight_kernel<<<(DIM   * GK + TPB - 1) / TPB, TPB, 0, stream>>>(W_out,  Bt_out,  DIM);

  // 1) v = value @ W_v + b_v          (86016 x 256 x 256)
  wmma_gemm_bias_kernel<<<(BS * LV) / 16, TPB, 0, stream>>>(value, Bt_v, b_v, v, DIM);

  // 2) off = query @ W_off + b_off    (32768 x 192 x 256)
  wmma_gemm_bias_kernel<<<(BS * LQ) / 16, TPB, 0, stream>>>(query, Bt_off, b_off, off, Noff);

  // 3) attn logits = query @ W_attn   (32768 x 96 x 256)
  wmma_gemm_bias_kernel<<<(BS * LQ) / 16, TPB, 0, stream>>>(query, Bt_attn, b_attn, attn, Nattn);

  // 4) softmax over 12 per (b,q,h), in place
  {
    long n = (long)BS * LQ * NH;
    softmax12_kernel<<<(int)((n + TPB - 1) / TPB), TPB, 0, stream>>>(attn);
  }

  // 5) bilinear sampling + attention-weighted sum -> samp (32768 x 256)
  {
    long waves = (long)BS * LQ * NH;
    deform_sample_kernel<<<(int)((waves * 32 + TPB - 1) / TPB), TPB, 0, stream>>>(
        v, off, attn, refpts, samp);
  }

  // 6) out = samp @ W_out + b_out     (32768 x 256 x 256)
  wmma_gemm_bias_kernel<<<(BS * LQ) / 16, TPB, 0, stream>>>(samp, Bt_out, b_out, outp, DIM);
}
